// DLRM_61134564491288
// MI455X (gfx1250) — matrix operand order repaired and checked by reading, not compiled, not run on hardware
//
#include <hip/hip_runtime.h>
#include <hip/hip_bf16.h>
#include <math.h>

// ---------------------------------------------------------------------------
// Types for WMMA fragments (CDNA5 wave32, v_wmma_f32_16x16x32_f16)
// ---------------------------------------------------------------------------
typedef __attribute__((ext_vector_type(16))) _Float16 v16h;
typedef __attribute__((ext_vector_type(8)))  float    v8f;
typedef __attribute__((ext_vector_type(8)))  unsigned short v8us;
typedef __attribute__((ext_vector_type(4)))  int      v4i;

#define B_ROWS   4096
#define HIST     50
#define EMB_DIM  64
#define ALL_DIM  192
#define TRIU_N   18528              // 192*193/2
#define OUT_K    18592              // TRIU_N + 64
#define BN_EPS   1e-5f

#if __has_builtin(__builtin_amdgcn_global_load_async_to_lds_b128)
#define HAVE_ASYNC_LDS 1
// Builtin expects: (v4i addrspace(1)* src, v4i addrspace(3)* dst, imm offset, imm cpol)
typedef __attribute__((address_space(1))) v4i gas_v4i;
typedef __attribute__((address_space(3))) v4i las_v4i;
#endif

__device__ __forceinline__ void wait_asynccnt0() {
#if defined(HAVE_ASYNC_LDS)
#if __has_builtin(__builtin_amdgcn_s_wait_asynccnt)
    __builtin_amdgcn_s_wait_asynccnt(0);
#else
    asm volatile("s_wait_asynccnt 0" ::: "memory");
#endif
#endif
}

// ---------------------------------------------------------------------------
// Weight transpose + f32 -> f16 convert:  W[K][N] -> Wt[N][K]  (f16)
// ---------------------------------------------------------------------------
__global__ __launch_bounds__(256) void wtrans_kernel(const float* __restrict__ W,
                                                     _Float16* __restrict__ Wt,
                                                     int K, int N) {
    __shared__ float tile[32][33];
    int k0 = blockIdx.x * 32, n0 = blockIdx.y * 32;
    for (int r = threadIdx.y; r < 32; r += 8) {
        int k = k0 + r, n = n0 + threadIdx.x;
        if (k < K && n < N) tile[r][threadIdx.x] = W[(size_t)k * N + n];
    }
    __syncthreads();
    for (int r = threadIdx.y; r < 32; r += 8) {
        int n = n0 + r, k = k0 + threadIdx.x;
        if (n < N && k < K) Wt[(size_t)n * K + k] = (_Float16)tile[threadIdx.x][r];
    }
}

// ---------------------------------------------------------------------------
// Triu pair table: pairs[k] = i | (j<<8).  k in [0,18528): triu(i,j) of 192;
// k in [18528,18592): i = k-18528 (bottom), j = 192 (the "1.0f" LDS slot).
// ---------------------------------------------------------------------------
__global__ void pairs_kernel(unsigned short* __restrict__ pairs) {
    int k = blockIdx.x * 256 + threadIdx.x;
    if (k >= OUT_K) return;
    int i, j;
    if (k < TRIU_N) {
        int rem = k; i = 0;
        while (rem >= ALL_DIM - i) { rem -= ALL_DIM - i; ++i; }
        j = i + rem;
    } else {
        i = k - TRIU_N; j = ALL_DIM;   // multiply by fs[m][192] == 1.0f
    }
    pairs[k] = (unsigned short)(i | (j << 8));
}

// ---------------------------------------------------------------------------
// Embedding gather + sum over history -> concat_first cols [128,192)
// ---------------------------------------------------------------------------
__global__ __launch_bounds__(64) void embed_kernel(const int* __restrict__ xs,
                                                   const float* __restrict__ emb,
                                                   float* __restrict__ CF) {
    int b = blockIdx.x, d = threadIdx.x;
    __shared__ int idx[HIST];
    if (d < HIST) idx[d] = xs[b * HIST + d];
    __syncthreads();
    float s = 0.f;
    #pragma unroll 5
    for (int h = 0; h < HIST; ++h) s += emb[(size_t)idx[h] * EMB_DIM + d];
    CF[(size_t)b * ALL_DIM + 128 + d] = s;
}

// ---------------------------------------------------------------------------
// Generic WMMA GEMM:  C[M,N](f32) = act( A[M,K](f32) @ Bt[N,K](f16)^T + bias )
// Block = 128 threads (4 waves). Wave tile = 32M x 64N (2 A frags x 4 B frags
// -> 8 WMMA per K-step, B reused across both A fragments).
// Block tile = 128M x 64N. Requires M%128==0, N%64==0, K%32==0.
// ---------------------------------------------------------------------------
__global__ __launch_bounds__(128) void gemm_f16_kernel(
    const float* __restrict__ A, int lda,
    const _Float16* __restrict__ Bt, int K,
    const float* __restrict__ bias,
    float* __restrict__ C, int ldc, int relu)
{
    const int lane  = threadIdx.x & 31;
    const int wave  = threadIdx.x >> 5;
    const int mWave = blockIdx.x * 128 + wave * 32;
    const int nb    = blockIdx.y * 64;
    const int lm    = lane & 15;
    const int koff  = (lane < 16) ? 0 : 16;   // A K sub-offset (contiguous k = 16h + s)
    const int kst   = (lane < 16) ? 0 : 16;   // B K sub-offset (contiguous k = 16h + s, same as A)
    const int nlane = nb + lm;
    const int m0    = mWave + lm;
    const int m1    = m0 + 16;

    v8f acc[2][4] = {{v8f{}, v8f{}, v8f{}, v8f{}}, {v8f{}, v8f{}, v8f{}, v8f{}}};

    for (int kb = 0; kb < K; kb += 32) {
        // Two A fragments (rows m0, m1): 16x32 f16 each, contiguous k = 16h + s per lane half
        const float* pa0 = A + (size_t)m0 * lda + kb + koff;
        const float* pa1 = A + (size_t)m1 * lda + kb + koff;
        v16h a0, a1;
        #pragma unroll
        for (int e = 0; e < 8; ++e) {
            a0[e]     = (_Float16)pa0[e];
            a0[8 + e] = (_Float16)pa0[8 + e];
            a1[e]     = (_Float16)pa1[e];
            a1[8 + e] = (_Float16)pa1[8 + e];
        }
        #pragma unroll
        for (int t = 0; t < 4; ++t) {
            const _Float16* pb = Bt + (size_t)(nlane + t * 16) * K + kb + kst;
            __builtin_prefetch(pb + 256, 0, 1);   // ~8 K-steps ahead -> global_prefetch_b8
            v16h b = *(const v16h*)pb;
            acc[0][t] = __builtin_amdgcn_wmma_f32_16x16x32_f16(
                false, a0, false, b, (short)0, acc[0][t], false, false);
            acc[1][t] = __builtin_amdgcn_wmma_f32_16x16x32_f16(
                false, a1, false, b, (short)0, acc[1][t], false, false);
        }
    }

    const int mrow = mWave + ((lane < 16) ? 0 : 8);
    #pragma unroll
    for (int t = 0; t < 4; ++t) {
        int n = nlane + t * 16;
        float bv = bias ? bias[n] : 0.f;
        #pragma unroll
        for (int s = 0; s < 2; ++s) {
            #pragma unroll
            for (int r = 0; r < 8; ++r) {
                float v = acc[s][t][r] + bv;
                if (relu) v = fmaxf(v, 0.f);
                C[(size_t)(mrow + s * 16 + r) * ldc + n] = v;
            }
        }
    }
}

// ---------------------------------------------------------------------------
// Fused top-MLP layer 1: t1 = relu( interaction(concat_first) @ tw1 + tb1 )
// Interaction features built on the fly from an LDS-staged concat_first tile
// (staged via CDNA5 async global->LDS when available).
// Wave tile = 16M x 128N (8 WMMA per K-step amortizing A construction).
// Block tile = 64M x 128N, grid = (4096/64, 1024/128).
// ---------------------------------------------------------------------------
#define FSS 200   // row stride for fs[] (192 features + slot 192 = 1.0f)

__global__ __launch_bounds__(128) void top1_kernel(
    const float* __restrict__ CF,                 // 4096 x 192
    const _Float16* __restrict__ Bt,              // 1024 x 18592 (tw1^T f16)
    const unsigned short* __restrict__ pairs,     // 18592, i|(j<<8)
    const float* __restrict__ bias,               // 1024
    float* __restrict__ C)                        // 4096 x 1024
{
    __shared__ float fs[64][FSS];
    const int tid   = threadIdx.x;
    const int mBase = blockIdx.x * 64;
    const int nb    = blockIdx.y * 128;

#if defined(HAVE_ASYNC_LDS)
    // Async DMA the 64x192 f32 tile straight into LDS (no VGPR round-trip).
    for (int idx = tid * 4; idx < 64 * ALL_DIM; idx += 128 * 4) {
        int r = idx / ALL_DIM, c = idx - r * ALL_DIM;      // c % 4 == 0
        const float* g = &CF[(size_t)(mBase + r) * ALL_DIM + c];
        float* l = &fs[r][c];
        __builtin_amdgcn_global_load_async_to_lds_b128((gas_v4i*)g, (las_v4i*)l, 0, 0);
    }
    wait_asynccnt0();
#else
    for (int idx = tid; idx < 64 * ALL_DIM; idx += 128) {
        int r = idx / ALL_DIM, c = idx - r * ALL_DIM;
        fs[r][c] = CF[(size_t)(mBase + r) * ALL_DIM + c];
    }
#endif
    if (tid < 64) fs[tid][ALL_DIM] = 1.0f;
    __syncthreads();

    const int lane  = tid & 31;
    const int wave  = tid >> 5;
    const int lm    = (wave << 4) + (lane & 15);   // local row 0..63
    const int koff  = (lane < 16) ? 0 : 16;
    const int kst   = (lane < 16) ? 0 : 16;
    const int nlane = nb + (lane & 15);

    v8f acc[8] = {v8f{}, v8f{}, v8f{}, v8f{}, v8f{}, v8f{}, v8f{}, v8f{}};

    for (int kb = 0; kb < OUT_K; kb += 32) {
        v8us p0 = *(const v8us*)(pairs + kb + koff);        // 16B aligned
        v8us p1 = *(const v8us*)(pairs + kb + 8 + koff);
        v16h a;
        #pragma unroll
        for (int e = 0; e < 8; ++e) {
            int pi0 = p0[e] & 255, pj0 = p0[e] >> 8;
            int pi1 = p1[e] & 255, pj1 = p1[e] >> 8;
            a[e]     = (_Float16)(fs[lm][pi0] * fs[lm][pj0]);
            a[8 + e] = (_Float16)(fs[lm][pi1] * fs[lm][pj1]);
        }
        #pragma unroll
        for (int t = 0; t < 8; ++t) {
            const _Float16* pb = Bt + (size_t)(nlane + t * 16) * OUT_K + kb + kst;
            __builtin_prefetch(pb + 256, 0, 1);
            v16h b = *(const v16h*)pb;
            acc[t] = __builtin_amdgcn_wmma_f32_16x16x32_f16(
                false, a, false, b, (short)0, acc[t], false, false);
        }
    }

    const int mrow = mBase + (wave << 4) + ((lane < 16) ? 0 : 8);
    #pragma unroll
    for (int t = 0; t < 8; ++t) {
        int n = nlane + t * 16;
        float bv = bias[n];
        #pragma unroll
        for (int r = 0; r < 8; ++r) {
            float v = fmaxf(acc[t][r] + bv, 0.f);
            C[(size_t)(mrow + r) * 1024 + n] = v;
        }
    }
}

// ---------------------------------------------------------------------------
// BatchNorm over batch dim (population stats), 64 features
// ---------------------------------------------------------------------------
__global__ __launch_bounds__(256) void bn_stats_kernel(const float* __restrict__ z,
                                                       float* __restrict__ stats) {
    int f = blockIdx.x, t = threadIdx.x;
    float s = 0.f, ss = 0.f;
    for (int b = t; b < B_ROWS; b += 256) {
        float v = z[(size_t)b * 64 + f];
        s += v; ss += v * v;
    }
    __shared__ float sh[256], sh2[256];
    sh[t] = s; sh2[t] = ss; __syncthreads();
    for (int o = 128; o > 0; o >>= 1) {
        if (t < o) { sh[t] += sh[t + o]; sh2[t] += sh2[t + o]; }
        __syncthreads();
    }
    if (t == 0) {
        float mu = sh[0] / (float)B_ROWS;
        stats[f]      = mu;
        stats[64 + f] = sh2[0] / (float)B_ROWS - mu * mu;
    }
}

__global__ __launch_bounds__(256) void bn_apply_kernel(
    const float* __restrict__ z, const float* __restrict__ stats,
    const float* __restrict__ gamma, const float* __restrict__ beta,
    float* __restrict__ CF, float* __restrict__ out_xembed) {
    int i = blockIdx.x * 256 + threadIdx.x;
    if (i >= B_ROWS * 64) return;
    int b = i >> 6, f = i & 63;
    float x = gamma[f] * (z[i] - stats[f]) * rsqrtf(stats[64 + f] + BN_EPS) + beta[f];
    CF[(size_t)b * ALL_DIM + 64 + f] = x;
    out_xembed[i] = x;
}

// ---------------------------------------------------------------------------
// Final layer: out[b] = sigmoid( dot(t2[b,:512], tw3) + tb3 )
// ---------------------------------------------------------------------------
__global__ __launch_bounds__(128) void top3_kernel(const float* __restrict__ t2,
                                                   const float* __restrict__ w,
                                                   const float* __restrict__ b3,
                                                   float* __restrict__ out) {
    int b = blockIdx.x, t = threadIdx.x;
    float s = 0.f;
    for (int k = t; k < 512; k += 128) s += t2[(size_t)b * 512 + k] * w[k];
    __shared__ float sh[128];
    sh[t] = s; __syncthreads();
    for (int o = 64; o > 0; o >>= 1) {
        if (t < o) sh[t] += sh[t + o];
        __syncthreads();
    }
    if (t == 0) out[b] = 1.f / (1.f + expf(-(sh[0] + b3[0])));
}

// ---------------------------------------------------------------------------
// Host launch
// ---------------------------------------------------------------------------
static inline size_t alignup(size_t x) { return (x + 255) & ~(size_t)255; }

extern "C" void kernel_launch(void* const* d_in, const int* in_sizes, int n_in,
                              void* d_out, int out_size, void* d_ws, size_t ws_size,
                              hipStream_t stream) {
    const int*   x_sparse = (const int*)  d_in[0];
    const float* x_dense  = (const float*)d_in[1];
    const float* x_ebp    = (const float*)d_in[2];
    const float* emb      = (const float*)d_in[3];
    const float* pw1 = (const float*)d_in[4];  const float* pb1 = (const float*)d_in[5];
    const float* pw2 = (const float*)d_in[6];  const float* pb2 = (const float*)d_in[7];
    const float* bng = (const float*)d_in[8];  const float* bnb = (const float*)d_in[9];
    const float* bw1 = (const float*)d_in[10]; const float* bb1 = (const float*)d_in[11];
    const float* bw2 = (const float*)d_in[12]; const float* bb2 = (const float*)d_in[13];
    const float* tw1 = (const float*)d_in[14]; const float* tb1 = (const float*)d_in[15];
    const float* tw2 = (const float*)d_in[16]; const float* tb2 = (const float*)d_in[17];
    const float* tw3 = (const float*)d_in[18]; const float* tb3 = (const float*)d_in[19];

    float* out_prob   = (float*)d_out;            // [4096]
    float* out_xembed = (float*)d_out + B_ROWS;   // [4096*64]

    // Workspace carve-up
    char* ws = (char*)d_ws; size_t off = 0;
    auto take = [&](size_t bytes) { char* p = ws + off; off = alignup(off + bytes); return p; };
    _Float16* wt_pw1 = (_Float16*)take((size_t)256 * 768  * 2);
    _Float16* wt_pw2 = (_Float16*)take((size_t)64  * 256  * 2);
    _Float16* wt_bw1 = (_Float16*)take((size_t)512 * 256  * 2);
    _Float16* wt_bw2 = (_Float16*)take((size_t)64  * 512  * 2);
    _Float16* wt_tw2 = (_Float16*)take((size_t)512 * 1024 * 2);
    _Float16* wt_tw1 = (_Float16*)take((size_t)1024 * OUT_K * 2);
    unsigned short* pairs = (unsigned short*)take((size_t)OUT_K * 2);
    float* h1    = (float*)take((size_t)B_ROWS * 256 * 4);
    float* z     = (float*)take((size_t)B_ROWS * 64  * 4);
    float* stats = (float*)take((size_t)128 * 4);
    float* bot_h = (float*)take((size_t)B_ROWS * 512 * 4);
    float* CF    = (float*)take((size_t)B_ROWS * ALL_DIM * 4);
    float* t1    = (float*)take((size_t)B_ROWS * 1024 * 4);
    float* t2    = (float*)take((size_t)B_ROWS * 512  * 4);
    (void)ws_size; (void)in_sizes; (void)n_in; (void)out_size;

    dim3 tb(32, 8);
    // Weight prep (f32 -> f16, transposed to N x K)
    wtrans_kernel<<<dim3(768/32,  256/32),  tb, 0, stream>>>(pw1, wt_pw1, 768,  256);
    wtrans_kernel<<<dim3(256/32,  64/32),   tb, 0, stream>>>(pw2, wt_pw2, 256,  64);
    wtrans_kernel<<<dim3(256/32,  512/32),  tb, 0, stream>>>(bw1, wt_bw1, 256,  512);
    wtrans_kernel<<<dim3(512/32,  64/32),   tb, 0, stream>>>(bw2, wt_bw2, 512,  64);
    wtrans_kernel<<<dim3(1024/32, 512/32),  tb, 0, stream>>>(tw2, wt_tw2, 1024, 512);
    wtrans_kernel<<<dim3(OUT_K/32,1024/32), tb, 0, stream>>>(tw1, wt_tw1, OUT_K,1024);
    pairs_kernel<<<(OUT_K + 255)/256, 256, 0, stream>>>(pairs);

    // Embedding gather-sum -> concat_first[:,128:192]
    embed_kernel<<<B_ROWS, 64, 0, stream>>>(x_sparse, emb, CF);

    // Projection MLP: h1 = relu(x_ebp @ pw1 + pb1); z = h1 @ pw2 + pb2
    gemm_f16_kernel<<<dim3(B_ROWS/128, 256/64), 128, 0, stream>>>(x_ebp, 768, wt_pw1, 768,  pb1, h1, 256, 1);
    gemm_f16_kernel<<<dim3(B_ROWS/128, 64/64),  128, 0, stream>>>(h1,    256, wt_pw2, 256,  pb2, z,  64,  0);
    // BatchNorm -> concat_first[:,64:128] and second output
    bn_stats_kernel<<<64, 256, 0, stream>>>(z, stats);
    bn_apply_kernel<<<(B_ROWS*64)/256, 256, 0, stream>>>(z, stats, bng, bnb, CF, out_xembed);

    // Bottom MLP -> concat_first[:,0:64]
    gemm_f16_kernel<<<dim3(B_ROWS/128, 512/64), 128, 0, stream>>>(x_dense, 256, wt_bw1, 256, bb1, bot_h, 512, 1);
    gemm_f16_kernel<<<dim3(B_ROWS/128, 64/64),  128, 0, stream>>>(bot_h,   512, wt_bw2, 512, bb2, CF, ALL_DIM, 1);

    // Fused interaction + top layer 1 (dominant GEMM, 156 GFLOP)
    top1_kernel<<<dim3(B_ROWS/64, 1024/128), 128, 0, stream>>>(CF, wt_tw1, pairs, tb1, t1);

    // Top layers 2 & 3
    gemm_f16_kernel<<<dim3(B_ROWS/128, 512/64), 128, 0, stream>>>(t1, 1024, wt_tw2, 1024, tb2, t2, 512, 1);
    top3_kernel<<<B_ROWS, 128, 0, stream>>>(t2, tw3, tb3, out_prob);
}
